// MultiheadAttention_59012850647137
// MI455X (gfx1250) — matrix operand order repaired and checked by reading, not compiled, not run on hardware
//
#include <hip/hip_runtime.h>
#include <hip/hip_bf16.h>
#include <stdint.h>

// ---------------------------------------------------------------------------
// MI455X (gfx1250) threshold-pruned multi-head attention.
// B=2, S=2048, D=1024, H=16, HD=64.  All GEMMs via v_wmma_f32_16x16x32_f16.
// Flash-style 3-pass attention; 512MB logits tensor never materialized.
// Round 4: interleave next-chunk B-fragment loads between current-chunk WMMA
// pairs so each consumed fragment dies before the next one goes live --
// drops peak VGPR pressure (~234 -> ~150) and removes the accumulator
// v_mov_b64 evictions the allocator was forced into.
// ---------------------------------------------------------------------------

typedef _Float16 v16h __attribute__((ext_vector_type(16)));
typedef _Float16 v4h  __attribute__((ext_vector_type(4)));
typedef float    v8f  __attribute__((ext_vector_type(8)));

#define BATCH  2
#define S_LEN  2048
#define DMODEL 1024
#define NHEAD  16
#define HDIM   64

union FragU { v16h v; uint4 q[2]; };

// A-operand (16x32 f16, M x K). Contiguous-K layout (not the ISA table order): lane m=lane&15, h=lane>>4;
// halves 0..15 <- K = 16h..16h+15.
__device__ __forceinline__ v16h load_frag_a(const _Float16* __restrict__ base, int ld) {
    const int lane = threadIdx.x & 31;
    const int m = lane & 15, hl = lane >> 4;
    FragU u;
    const _Float16* p = base + m * ld + 16 * hl;
    u.q[0] = *(const uint4*)(p);
    u.q[1] = *(const uint4*)(p + 8);
    return u.v;
}

// B-operand (32x16 f16, K x N) loaded from B^T rows (row n, contiguous K).
// Same contiguous-K order as A: lane n=lane&15, h=lane>>4; halves 0..15 <- K = 16h..16h+15.
__device__ __forceinline__ v16h load_frag_bt(const _Float16* __restrict__ base, int ld) {
    const int lane = threadIdx.x & 31;
    const int n = lane & 15, hl = lane >> 4;
    FragU u;
    const _Float16* p = base + n * ld + 16 * hl;
    u.q[0] = *(const uint4*)(p);
    u.q[1] = *(const uint4*)(p + 8);
    return u.v;
}

__device__ __forceinline__ v8f wmma_f16(v16h a, v16h b, v8f c) {
    return __builtin_amdgcn_wmma_f32_16x16x32_f16(false, a, false, b, (short)0, c, false, false);
}

// Reductions over the 16-lane half-wave (xor masks 1..8 stay within a half).
__device__ __forceinline__ float half_max(float v) {
    for (int msk = 8; msk; msk >>= 1) v = fmaxf(v, __shfl_xor(v, msk, 32));
    return v;
}
__device__ __forceinline__ float half_sum(float v) {
    for (int msk = 8; msk; msk >>= 1) v += __shfl_xor(v, msk, 32);
    return v;
}

// ---------------------------------------------------------------------------
// Vectorized f32 -> f16 convert (all arrays are multiples of 4 elements).
__global__ void f32_to_f16_kernel(const float* __restrict__ s, _Float16* __restrict__ d, int n4) {
    int i = blockIdx.x * blockDim.x + threadIdx.x;
    const int stride = gridDim.x * blockDim.x;
    for (; i < n4; i += stride) {
        float4 f = ((const float4*)s)[i];
        v4h h = { (_Float16)f.x, (_Float16)f.y, (_Float16)f.z, (_Float16)f.w };
        ((v4h*)d)[i] = h;
    }
}

// ---------------------------------------------------------------------------
// GEMM pipeline helpers: 32(M) x 64(N) wave tile, K chunk of 32.
// chunk_step: run the WMMAs of the "current" fragment set while loading the
// "next" set, with each next-B load issued right before the current-B frag it
// will replace is consumed (minimizes simultaneous fragment liveness).
__device__ __forceinline__ void load_chunk(v16h& a0, v16h& a1, v16h (&b)[4],
                                           const _Float16* __restrict__ Ab,
                                           const _Float16* __restrict__ Wb, int kc) {
    const int nk = kc * 32;
    a0 = load_frag_a(Ab + nk, DMODEL);
    a1 = load_frag_a(Ab + 16 * DMODEL + nk, DMODEL);
#pragma unroll
    for (int dt = 0; dt < 4; ++dt)
        b[dt] = load_frag_bt(Wb + (size_t)dt * 16 * DMODEL + nk, DMODEL);
}

__device__ __forceinline__ void chunk_step(v8f (&acc)[2][4],
                                           const v16h& a0c, const v16h& a1c, const v16h (&bc)[4],
                                           v16h& a0n, v16h& a1n, v16h (&bn)[4],
                                           const _Float16* __restrict__ Ab,
                                           const _Float16* __restrict__ Wb, int kn) {
    const int nk = kn * 32;
    a0n = load_frag_a(Ab + nk, DMODEL);
    a1n = load_frag_a(Ab + 16 * DMODEL + nk, DMODEL);
#pragma unroll
    for (int dt = 0; dt < 4; ++dt) {
        bn[dt] = load_frag_bt(Wb + (size_t)dt * 16 * DMODEL + nk, DMODEL);
        acc[0][dt] = wmma_f16(a0c, bc[dt], acc[0][dt]);
        acc[1][dt] = wmma_f16(a1c, bc[dt], acc[1][dt]);
    }
}

__device__ __forceinline__ void chunk_mma(v8f (&acc)[2][4],
                                          const v16h& a0c, const v16h& a1c, const v16h (&bc)[4]) {
#pragma unroll
    for (int dt = 0; dt < 4; ++dt) {
        acc[0][dt] = wmma_f16(a0c, bc[dt], acc[0][dt]);
        acc[1][dt] = wmma_f16(a1c, bc[dt], acc[1][dt]);
    }
}

// ---------------------------------------------------------------------------
// QKV GEMM: out[row,e] = sum_d x[row,d]*qkv_w[e,d] + qkv_b[e]
// Grid: (M/32, 3D/64). 64-wide N blocks align exactly with per-head q/k/v
// segments: sel = blockIdx.y%3, head = blockIdx.y/3.
// Q,K stored [b,h,s,hd] f16; V stored transposed [b,h,hd,s] f16 so the
// attention A*V B-operands are contiguous.
__global__ __launch_bounds__(32)
void qkv_gemm_kernel(const _Float16* __restrict__ X, const _Float16* __restrict__ W,
                     const float* __restrict__ bias,
                     _Float16* __restrict__ Q, _Float16* __restrict__ K,
                     _Float16* __restrict__ VT) {
    const int mt = blockIdx.x, nt = blockIdx.y;
    const int lane = threadIdx.x & 31;
    const int n = lane & 15, hl = lane >> 4;
    const _Float16* Xb = X + (size_t)mt * 32 * DMODEL;
    const _Float16* Wb = W + (size_t)nt * 64 * DMODEL;
    const int NC = DMODEL / 32;

    v8f acc[2][4] = {};
    v16h a0A, a1A, a0B, a1B;
    v16h bA[4], bB[4];
    load_chunk(a0A, a1A, bA, Xb, Wb, 0);
    for (int kc = 0; kc < NC - 2; kc += 2) {
        chunk_step(acc, a0A, a1A, bA, a0B, a1B, bB, Xb, Wb, kc + 1); // mma kc, ld kc+1
        chunk_step(acc, a0B, a1B, bB, a0A, a1A, bA, Xb, Wb, kc + 2); // mma kc+1, ld kc+2
    }
    chunk_step(acc, a0A, a1A, bA, a0B, a1B, bB, Xb, Wb, NC - 1);     // mma NC-2, ld NC-1
    chunk_mma(acc, a0B, a1B, bB);                                    // mma NC-1

    const int sel = nt % 3, head = nt / 3;
#pragma unroll
    for (int mh = 0; mh < 2; ++mh) {
#pragma unroll
        for (int dt = 0; dt < 4; ++dt) {
#pragma unroll
            for (int r = 0; r < 8; ++r) {
                const int m = r + 8 * hl;
                const int row = mt * 32 + mh * 16 + m;
                const int b = row >> 11, s = row & (S_LEN - 1);
                const int c = dt * 16 + n;
                const int e = nt * 64 + c;
                float v = acc[mh][dt][r] + bias[e];
                _Float16 hv = (_Float16)v;
                const size_t bh = (size_t)(b * NHEAD + head);
                if (sel == 0)      Q[(bh * S_LEN + s) * HDIM + c] = hv;
                else if (sel == 1) K[(bh * S_LEN + s) * HDIM + c] = hv;
                else               VT[(bh * HDIM + c) * S_LEN + s] = hv;
            }
        }
    }
}

// ---------------------------------------------------------------------------
// Attention: one wave per (b,h, 16-row q tile). Three passes over K:
//  P1: online rowmax M, rowsum Z of exp(l-M)      (flash softmax stats)
//  P2: cutoff = M + log(t*Z); Z2 = sum_{l>cut} exp(l-M)
//  P3: O = sum w*V (w via LDS transpose -> A-frag), then O /= Z2.
// Degenerate all-masked row (reference -> uniform 1/S): Z2==0 -> w=1, Z2=S.
// K-tile fragments ping-pong double-buffered in all passes.
__global__ __launch_bounds__(32)
void attn_kernel(const _Float16* __restrict__ Q, const _Float16* __restrict__ K,
                 const _Float16* __restrict__ VT, const float* __restrict__ thr,
                 _Float16* __restrict__ vals) {
    const int qt = blockIdx.x;         // q tile (0..127)
    const int bh = blockIdx.y;         // 0..31
    const int b = bh >> 4, h = bh & 15;
    const int lane = threadIdx.x & 31;
    const int n = lane & 15, hl = lane >> 4;
    const float t = thr[h];
    const float scale = 0.125f;        // 1/sqrt(64)
    const int NKT = S_LEN / 16;        // 128 k tiles

    const _Float16* Qb = Q + ((size_t)bh * S_LEN + qt * 16) * HDIM;
    const _Float16* Kb = K + (size_t)bh * S_LEN * HDIM;
    const _Float16* Vb = VT + (size_t)bh * HDIM * S_LEN;

    const v16h a0 = load_frag_a(Qb, HDIM);       // q cols 0..31
    const v16h a1 = load_frag_a(Qb + 32, HDIM);  // q cols 32..63

    float M[8], Z[8];
#pragma unroll
    for (int r = 0; r < 8; ++r) { M[r] = -INFINITY; Z[r] = 0.f; }

    // logits tile from a K-tile fragment pair
    auto logits = [&](v16h kb0, v16h kb1) -> v8f {
        v8f acc = {};
        acc = wmma_f16(a0, kb0, acc);
        acc = wmma_f16(a1, kb1, acc);
        return acc;
    };

    // ---- Pass 1: softmax stats -------------------------------------------
    auto p1_update = [&](v8f acc) {
#pragma unroll
        for (int r = 0; r < 8; ++r) {
            float l = acc[r] * scale;
            float tm = half_max(l);
            float nM = fmaxf(M[r], tm);
            float ts = half_sum(__expf(l - nM));
            Z[r] = Z[r] * __expf(M[r] - nM) + ts;
            M[r] = nM;
        }
    };
    {
        v16h p0a = load_frag_bt(Kb, HDIM), p0b = load_frag_bt(Kb + 32, HDIM);
        v16h p1a, p1b;
        for (int kt = 0; kt < NKT - 2; kt += 2) {
            const _Float16* k1 = Kb + (size_t)(kt + 1) * 16 * HDIM;
            __builtin_prefetch(k1 + 32 * HDIM, 0, 0);   // global_prefetch ahead
            p1a = load_frag_bt(k1, HDIM);
            p1b = load_frag_bt(k1 + 32, HDIM);
            p1_update(logits(p0a, p0b));
            const _Float16* k2 = Kb + (size_t)(kt + 2) * 16 * HDIM;
            p0a = load_frag_bt(k2, HDIM);
            p0b = load_frag_bt(k2 + 32, HDIM);
            p1_update(logits(p1a, p1b));
        }
        const _Float16* kl = Kb + (size_t)(NKT - 1) * 16 * HDIM;
        p1a = load_frag_bt(kl, HDIM);
        p1b = load_frag_bt(kl + 32, HDIM);
        p1_update(logits(p0a, p0b));
        p1_update(logits(p1a, p1b));
    }

    // ---- Pass 2: masked partition function -------------------------------
    float cut[8], Z2[8];
#pragma unroll
    for (int r = 0; r < 8; ++r) { cut[r] = M[r] + __logf(t * Z[r]); Z2[r] = 0.f; }
    auto p2_update = [&](v8f acc) {
#pragma unroll
        for (int r = 0; r < 8; ++r) {
            float l = acc[r] * scale;
            float w = (l > cut[r]) ? __expf(l - M[r]) : 0.f;
            Z2[r] += half_sum(w);
        }
    };
    {
        v16h p0a = load_frag_bt(Kb, HDIM), p0b = load_frag_bt(Kb + 32, HDIM);
        v16h p1a, p1b;
        for (int kt = 0; kt < NKT - 2; kt += 2) {
            const _Float16* k1 = Kb + (size_t)(kt + 1) * 16 * HDIM;
            p1a = load_frag_bt(k1, HDIM);
            p1b = load_frag_bt(k1 + 32, HDIM);
            p2_update(logits(p0a, p0b));
            const _Float16* k2 = Kb + (size_t)(kt + 2) * 16 * HDIM;
            p0a = load_frag_bt(k2, HDIM);
            p0b = load_frag_bt(k2 + 32, HDIM);
            p2_update(logits(p1a, p1b));
        }
        const _Float16* kl = Kb + (size_t)(NKT - 1) * 16 * HDIM;
        p1a = load_frag_bt(kl, HDIM);
        p1b = load_frag_bt(kl + 32, HDIM);
        p2_update(logits(p0a, p0b));
        p2_update(logits(p1a, p1b));
    }
    float unif[8];
#pragma unroll
    for (int r = 0; r < 8; ++r) {
        unif[r] = (Z2[r] == 0.f) ? 1.f : 0.f;
        if (Z2[r] == 0.f) Z2[r] = (float)S_LEN;
    }

    // ---- Pass 3: O = sum w*V ---------------------------------------------
    __shared__ __align__(16) _Float16 ldsw[16 * 32];  // w tile, C->A transpose
    v8f o[4] = {};
    auto w_to_lds = [&](v8f acc, int par) {
#pragma unroll
        for (int r = 0; r < 8; ++r) {
            float l = acc[r] * scale;
            float w = (l > cut[r]) ? __expf(l - M[r]) : 0.f;
            w = unif[r] != 0.f ? 1.f : w;       // degenerate row -> uniform
            // C layout (m = r+8h in vgpr, n in lane) -> LDS row-major tile
            ldsw[(r + 8 * hl) * 32 + par * 16 + n] = (_Float16)w;
        }
    };
    {
        v16h p0a = load_frag_bt(Kb, HDIM), p0b = load_frag_bt(Kb + 32, HDIM);
        v16h p1a, p1b;
        for (int cc = 0; cc < S_LEN / 32; ++cc) {
            const _Float16* k1 = Kb + (size_t)(2 * cc + 1) * 16 * HDIM;
            p1a = load_frag_bt(k1, HDIM);
            p1b = load_frag_bt(k1 + 32, HDIM);
            w_to_lds(logits(p0a, p0b), 0);
            const _Float16* k2 = Kb + (size_t)((2 * cc + 2) & (NKT - 1)) * 16 * HDIM;
            p0a = load_frag_bt(k2, HDIM);      // next cc's first tile (wraps)
            p0b = load_frag_bt(k2 + 32, HDIM);
            w_to_lds(logits(p1a, p1b), 1);
            __syncthreads();
            v16h aw = load_frag_a(ldsw, 32);   // A-frag of w, K-dim = 32 keys
#pragma unroll
            for (int dt = 0; dt < 4; ++dt) {
                v16h bv = load_frag_bt(Vb + (size_t)dt * 16 * S_LEN + cc * 32, S_LEN);
                o[dt] = wmma_f16(aw, bv, o[dt]);
            }
            __syncthreads();
        }
    }

    // ---- normalize + store into [B*S, D] f16 -----------------------------
#pragma unroll
    for (int dt = 0; dt < 4; ++dt) {
#pragma unroll
        for (int r = 0; r < 8; ++r) {
            const int m = r + 8 * hl;
            const size_t row = (size_t)b * S_LEN + qt * 16 + m;
            float v = o[dt][r] / Z2[r];
            vals[row * DMODEL + h * HDIM + dt * 16 + n] = (_Float16)v;
        }
    }
}

// ---------------------------------------------------------------------------
// O projection: out[row,e] = sum_d vals[row,d]*o_w[e,d] + o_b[e]  (f32 out)
// Same 32x64 interleaved-pipeline wave tile as the QKV GEMM.
__global__ __launch_bounds__(32)
void oproj_kernel(const _Float16* __restrict__ A, const _Float16* __restrict__ W,
                  const float* __restrict__ bias, float* __restrict__ out) {
    const int mt = blockIdx.x, nt = blockIdx.y;
    const int lane = threadIdx.x & 31;
    const int n = lane & 15, hl = lane >> 4;
    const _Float16* Ab = A + (size_t)mt * 32 * DMODEL;
    const _Float16* Wb = W + (size_t)nt * 64 * DMODEL;
    const int NC = DMODEL / 32;

    v8f acc[2][4] = {};
    v16h a0A, a1A, a0B, a1B;
    v16h bA[4], bB[4];
    load_chunk(a0A, a1A, bA, Ab, Wb, 0);
    for (int kc = 0; kc < NC - 2; kc += 2) {
        chunk_step(acc, a0A, a1A, bA, a0B, a1B, bB, Ab, Wb, kc + 1);
        chunk_step(acc, a0B, a1B, bB, a0A, a1A, bA, Ab, Wb, kc + 2);
    }
    chunk_step(acc, a0A, a1A, bA, a0B, a1B, bB, Ab, Wb, NC - 1);
    chunk_mma(acc, a0B, a1B, bB);

#pragma unroll
    for (int mh = 0; mh < 2; ++mh) {
#pragma unroll
        for (int dt = 0; dt < 4; ++dt) {
#pragma unroll
            for (int r = 0; r < 8; ++r) {
                const int m = r + 8 * hl;
                const size_t row = (size_t)mt * 32 + mh * 16 + m;
                const int e = nt * 64 + dt * 16 + n;
                out[row * DMODEL + e] = acc[mh][dt][r] + bias[e];
            }
        }
    }
}

// ---------------------------------------------------------------------------
extern "C" void kernel_launch(void* const* d_in, const int* in_sizes, int n_in,
                              void* d_out, int out_size, void* d_ws, size_t ws_size,
                              hipStream_t stream) {
    (void)in_sizes; (void)n_in; (void)out_size; (void)ws_size;
    const float* x      = (const float*)d_in[0];
    const float* thr    = (const float*)d_in[1];
    const float* qkv_w  = (const float*)d_in[2];
    const float* qkv_b  = (const float*)d_in[3];
    const float* o_w    = (const float*)d_in[4];
    const float* o_b    = (const float*)d_in[5];
    float* out = (float*)d_out;

    const int xN    = BATCH * S_LEN * DMODEL;   // 4194304
    const int qkvwN = 3 * DMODEL * DMODEL;      // 3145728
    const int owN   = DMODEL * DMODEL;          // 1048576
    const size_t hdN = (size_t)BATCH * NHEAD * S_LEN * HDIM;  // 4194304

    char* ws = (char*)d_ws;
    _Float16* xh    = (_Float16*)(ws);                    ws += (size_t)xN * 2;
    _Float16* qkvwh = (_Float16*)(ws);                    ws += (size_t)qkvwN * 2;
    _Float16* owh   = (_Float16*)(ws);                    ws += (size_t)owN * 2;
    _Float16* Qh    = (_Float16*)(ws);                    ws += hdN * 2;
    _Float16* Kh    = (_Float16*)(ws);                    ws += hdN * 2;
    _Float16* VTh   = (_Float16*)(ws);                    ws += hdN * 2;
    _Float16* vals  = (_Float16*)(ws);                    ws += (size_t)xN * 2;

    f32_to_f16_kernel<<<512, 256, 0, stream>>>(x, xh, xN / 4);
    f32_to_f16_kernel<<<512, 256, 0, stream>>>(qkv_w, qkvwh, qkvwN / 4);
    f32_to_f16_kernel<<<512, 256, 0, stream>>>(o_w, owh, owN / 4);

    qkv_gemm_kernel<<<dim3(BATCH * S_LEN / 32, 3 * DMODEL / 64), 32, 0, stream>>>(
        xh, qkvwh, qkv_b, Qh, Kh, VTh);

    attn_kernel<<<dim3(S_LEN / 16, BATCH * NHEAD), 32, 0, stream>>>(
        Qh, Kh, VTh, thr, vals);

    oproj_kernel<<<dim3(BATCH * S_LEN / 32, DMODEL / 64), 32, 0, stream>>>(
        vals, owh, o_b, out);
}
